// MixtureOfExperts_73727408603795
// MI455X (gfx1250) — matrix operand order repaired and checked by reading, not compiled, not run on hardware
//
#include <hip/hip_runtime.h>

// ---------------- problem constants ----------------
#define B_ 4
#define N_ 2048
#define D_ 512
#define E_ 16
#define H_ 1024
#define CAP_ 256            // N*2.0/E
#define BN_ (B_ * N_)       // 8192 tokens
#define SLOTS_ (E_ * B_ * CAP_)  // 16384 expert rows

typedef __attribute__((ext_vector_type(16))) __bf16 v16bf;
typedef __attribute__((ext_vector_type(8)))  float  v8f;

struct U32x8 { uint4 a; uint4 b; };

__device__ __forceinline__ unsigned short f2bf(float f) {
  unsigned u = __float_as_uint(f);
  unsigned r = ((u >> 16) & 1u) + 0x7fffu;   // round-to-nearest-even
  return (unsigned short)((u + r) >> 16);
}

// B-fragment (and generic): 16 contiguous bf16 (32B) from a [N][K] row
__device__ __forceinline__ v16bf ld16_contig(const unsigned short* p) {
  U32x8 t;
  t.a = *reinterpret_cast<const uint4*>(p);
  t.b = *reinterpret_cast<const uint4*>(p + 8);
  return __builtin_bit_cast(v16bf, t);
}

// A-fragment: per lane, 16 contiguous k at k+lh*16 (matches B contiguous order)
__device__ __forceinline__ v16bf ld16_afrag(const unsigned short* rowk, int lh) {
  const unsigned short* p = rowk + lh * 16;
  U32x8 t;
  t.a = *reinterpret_cast<const uint4*>(p);
  t.b = *reinterpret_cast<const uint4*>(p + 8);
  return __builtin_bit_cast(v16bf, t);
}

__device__ __forceinline__ v8f wmma_bf16(v16bf a, v16bf b, v8f c) {
  return __builtin_amdgcn_wmma_f32_16x16x32_bf16(
      /*neg_a=*/false, a, /*neg_b=*/false, b,
      /*c_mod=*/(short)0, c, /*reuse_a=*/false, /*reuse_b=*/false);
}

// ---------------- prep kernels ----------------
__global__ __launch_bounds__(256) void cvt_f32_bf16_kernel(
    const float* __restrict__ src, unsigned short* __restrict__ dst, int n) {
  int i = (blockIdx.x * 256 + threadIdx.x) * 4;
  if (i + 3 < n) {
    float4 v = *reinterpret_cast<const float4*>(src + i);
    ushort4 o;
    o.x = f2bf(v.x); o.y = f2bf(v.y); o.z = f2bf(v.z); o.w = f2bf(v.w);
    *reinterpret_cast<ushort4*>(dst + i) = o;
  }
}

// dst[c][r] = bf16(src[r][c]); batched over blockIdx.z
__global__ __launch_bounds__(256) void transpose_cvt_kernel(
    const float* __restrict__ src, unsigned short* __restrict__ dst,
    int R, int C, size_t sbatch, size_t dbatch) {
  __shared__ float tile[32][33];
  src += (size_t)blockIdx.z * sbatch;
  dst += (size_t)blockIdx.z * dbatch;
  int c0 = blockIdx.x * 32, r0 = blockIdx.y * 32;
  int tx = threadIdx.x, ty = threadIdx.y;   // (32, 8)
#pragma unroll
  for (int i = 0; i < 32; i += 8)
    tile[ty + i][tx] = src[(size_t)(r0 + ty + i) * C + (c0 + tx)];
  __syncthreads();
#pragma unroll
  for (int i = 0; i < 32; i += 8)
    dst[(size_t)(c0 + ty + i) * R + (r0 + tx)] = f2bf(tile[tx][ty + i]);
}

// ---------------- linear1: h = X @ W1 + b1 (WMMA bf16, f32 accum) ----------------
// grid (BN/64, D/128), 256 threads (8 waves). Wave: 4 M-tiles x 1 N-tile.
__global__ __launch_bounds__(256) void linear1_kernel(
    const unsigned short* __restrict__ Xbf,   // [BN][D] bf16
    const unsigned short* __restrict__ W1t,   // [D out][D in] bf16 (transposed)
    const float* __restrict__ b1,
    float* __restrict__ hf,                   // [BN][D] f32
    unsigned short* __restrict__ hb) {        // [BN][D] bf16
  const int lane = threadIdx.x & 31, wv = threadIdx.x >> 5;
  const int l15 = lane & 15, lh = lane >> 4;
  const int m0 = blockIdx.x * 64;
  const int ncol = blockIdx.y * 128 + wv * 16 + l15;
  const unsigned short* brow = W1t + (size_t)ncol * D_;
  v8f acc[4] = {v8f{}, v8f{}, v8f{}, v8f{}};
#pragma unroll 4
  for (int kc = 0; kc < D_; kc += 32) {
    v16bf bf = ld16_contig(brow + kc + lh * 16);
#pragma unroll
    for (int mt = 0; mt < 4; ++mt) {
      const unsigned short* arow = Xbf + (size_t)(m0 + mt * 16 + l15) * D_ + kc;
      acc[mt] = wmma_bf16(ld16_afrag(arow, lh), bf, acc[mt]);
    }
  }
  float bias = b1[ncol];
#pragma unroll
  for (int mt = 0; mt < 4; ++mt)
#pragma unroll
    for (int r = 0; r < 8; ++r) {
      int row = m0 + mt * 16 + lh * 8 + r;
      float v = acc[mt][r] + bias;
      hf[(size_t)row * D_ + ncol] = v;
      hb[(size_t)row * D_ + ncol] = f2bf(v);
    }
}

// ---------------- gating: softmax + top2 + random-second threshold ----------------
__global__ __launch_bounds__(256) void gating_kernel(
    const float* __restrict__ h, const float* __restrict__ Wg,
    const float* __restrict__ probs,
    int* __restrict__ idx1o, int* __restrict__ idx2o,
    float* __restrict__ g1o, float* __restrict__ g2o, int* __restrict__ keep2o) {
  int t = blockIdx.x * 256 + threadIdx.x;
  if (t >= BN_) return;
  const float* hr = h + (size_t)t * D_;
  float acc[E_];
#pragma unroll
  for (int e = 0; e < E_; ++e) acc[e] = 0.f;
  for (int d = 0; d < D_; ++d) {
    float x = hr[d];
    const float* w = Wg + (size_t)d * E_;
#pragma unroll
    for (int e = 0; e < E_; ++e) acc[e] += x * w[e];
  }
  float m = acc[0];
#pragma unroll
  for (int e = 1; e < E_; ++e) m = fmaxf(m, acc[e]);
  float s = 0.f;
#pragma unroll
  for (int e = 0; e < E_; ++e) { acc[e] = __expf(acc[e] - m); s += acc[e]; }
  float inv = 1.0f / s;
#pragma unroll
  for (int e = 0; e < E_; ++e) acc[e] *= inv;
  int i1 = 0; float g1 = acc[0];
#pragma unroll
  for (int e = 1; e < E_; ++e) if (acc[e] > g1) { g1 = acc[e]; i1 = e; }
  int i2 = (i1 == 0) ? 1 : 0; float g2 = acc[i2];
#pragma unroll
  for (int e = 0; e < E_; ++e)
    if (e != i1 && acc[e] > g2) { g2 = acc[e]; i2 = e; }
  float denom = g1 + g2 + 1e-9f;
  float g1n = g1 / denom, g2n = g2 / denom;
  idx1o[t] = i1; idx2o[t] = i2;
  g1o[t] = g1n; g2o[t] = g2n;
  keep2o[t] = (probs[t] < g2n * 5.0f) ? 1 : 0;   // second_threshold = 0.2
}

// ---------------- routing: ordered capacity scan (deterministic, no atomics) ----
__device__ __forceinline__ int block_scan_incl(int* sc, int tid, int flag) {
  sc[tid] = flag;
  __syncthreads();
#pragma unroll
  for (int off = 1; off < 256; off <<= 1) {
    int add = (tid >= off) ? sc[tid - off] : 0;
    __syncthreads();
    sc[tid] += add;
    __syncthreads();
  }
  return sc[tid];
}

__global__ __launch_bounds__(256) void routing_kernel(
    const int* __restrict__ idx1, const int* __restrict__ idx2,
    const int* __restrict__ keep2,
    int* __restrict__ slot_token, int* __restrict__ r1, int* __restrict__ r2) {
  __shared__ int sc[256];
  __shared__ int running;
  const int b = blockIdx.x >> 4;
  const int e = blockIdx.x & 15;
  const int tid = threadIdx.x;
  const int rowbase = (e * B_ + b) * CAP_;
  for (int c = tid; c < CAP_; c += 256) slot_token[rowbase + c] = -1;
  if (tid == 0) running = 0;
  __syncthreads();
  // phase 1: expert-1 assignments in token order
  for (int base = 0; base < N_; base += 256) {
    int tok = base + tid;
    int bt = b * N_ + tok;
    int sel = (idx1[bt] == e) ? 1 : 0;
    int inc = block_scan_incl(sc, tid, sel);
    int total = sc[255];
    int pos = running + inc - sel;
    if (sel) {
      if (pos < CAP_) { slot_token[rowbase + pos] = tok; r1[bt] = rowbase + pos; }
      else r1[bt] = -1;
    }
    __syncthreads();
    if (tid == 0) running += total;
    __syncthreads();
  }
  if (tid == 0) running = (running < CAP_) ? running : CAP_;  // mask1_count
  __syncthreads();
  // phase 2: expert-2, offset by mask1_count
  for (int base = 0; base < N_; base += 256) {
    int tok = base + tid;
    int bt = b * N_ + tok;
    int sel0 = (idx2[bt] == e) ? 1 : 0;
    int sel = sel0 & keep2[bt];
    int inc = block_scan_incl(sc, tid, sel);
    int total = sc[255];
    int pos = running + inc - sel;
    if (sel0) {
      if (sel && pos < CAP_) { slot_token[rowbase + pos] = tok; r2[bt] = rowbase + pos; }
      else r2[bt] = -1;
    }
    __syncthreads();
    if (tid == 0) running += total;
    __syncthreads();
  }
}

// ---------------- dispatch gather via the gfx1250 async-LDS DMA path ----------
// 4 slot-rows per block; each thread moves one 16B chunk:
//   global --(global_load_async_to_lds_b128)--> LDS stage
//   LDS stage --(global_store_async_from_lds_b128)--> exp_in
// Empty slots zero-fill their LDS chunk with a DS store instead.
// ASYNCcnt/DScnt ordering enforced with s_wait_* (wave-level).
__global__ __launch_bounds__(256) void dispatch_kernel(
    const int* __restrict__ slot_token, const unsigned short* __restrict__ hb,
    unsigned short* __restrict__ exp_in) {
  __shared__ __attribute__((aligned(16))) unsigned short stage[4 * D_];  // 4 KB
  const int tid = threadIdx.x;
  const int r = tid >> 6;                 // slot-row within block (64 chunks/row)
  const int c = tid & 63;                 // 16B chunk within the 1KB row
  const int s = blockIdx.x * 4 + r;
  const int tok = slot_token[s];
  const int b = (s >> 8) & (B_ - 1);
  const unsigned lds_off = (unsigned)tid * 16u;  // stage[] is the only LDS object
  if (tok >= 0) {
    const unsigned short* src = hb + (size_t)(b * N_ + tok) * D_ + c * 8;
    unsigned long long ga = (unsigned long long)(uintptr_t)src;
    asm volatile("global_load_async_to_lds_b128 %0, %1, off"
                 :: "v"(lds_off), "v"(ga) : "memory");
  } else {
    uint4 z = {0u, 0u, 0u, 0u};
    *reinterpret_cast<uint4*>(&stage[(size_t)tid * 8]) = z;   // DS zero-fill
  }
  asm volatile("s_wait_dscnt 0x0" ::: "memory");     // LDS zero-fills visible
  asm volatile("s_wait_asynccnt 0x0" ::: "memory");  // async loads landed in LDS
  {
    unsigned short* dst = exp_in + (size_t)s * D_ + c * 8;
    unsigned long long gd = (unsigned long long)(uintptr_t)dst;
    asm volatile("global_store_async_from_lds_b128 %0, %1, off"
                 :: "v"(gd), "v"(lds_off) : "memory");
  }
  asm volatile("s_wait_asynccnt 0x0" ::: "memory");  // drain before wave exit
}

// ---------------- fused expert FFN: leaky(exp_in@We1) @ We2, hid in LDS -------
// grid = E * 32 blocks (32 rows each), 256 threads / 8 waves.
__global__ __launch_bounds__(256) void expert_ffn_kernel(
    const unsigned short* __restrict__ exp_in,  // [E*1024][512] bf16
    const unsigned short* __restrict__ We1t,    // [E][H][D]  bf16 (col-major of We1)
    const unsigned short* __restrict__ We2t,    // [E][D][H]  bf16 (col-major of We2)
    float* __restrict__ exp_out) {              // [E*1024][512] f32
  __shared__ unsigned short hid[32 * H_];       // 64 KB bf16 tile
  const int e  = blockIdx.x >> 5;
  const int rb = blockIdx.x & 31;
  const int lane = threadIdx.x & 31, wv = threadIdx.x >> 5;
  const int l15 = lane & 15, lh = lane >> 4;
  const unsigned short* A  = exp_in + (size_t)(e * (B_ * CAP_) + rb * 32) * D_;
  const unsigned short* B1 = We1t + (size_t)e * H_ * D_;
  const unsigned short* B2 = We2t + (size_t)e * D_ * H_;

  // GEMM1: [32 x 512] x [512 x 1024] -> hid (leaky-relu, bf16, LDS)
  for (int nt = wv * 8; nt < wv * 8 + 8; ++nt) {
    v8f acc0 = {}, acc1 = {};
    const int ncol = nt * 16 + l15;
    const unsigned short* brow = B1 + (size_t)ncol * D_;
#pragma unroll 4
    for (int kc = 0; kc < D_; kc += 32) {
      v16bf bf = ld16_contig(brow + kc + lh * 16);
      acc0 = wmma_bf16(ld16_afrag(A + (size_t)(l15) * D_ + kc, lh), bf, acc0);
      acc1 = wmma_bf16(ld16_afrag(A + (size_t)(16 + l15) * D_ + kc, lh), bf, acc1);
    }
#pragma unroll
    for (int r = 0; r < 8; ++r) {
      float v0 = acc0[r]; v0 = (v0 > 0.f) ? v0 : 0.01f * v0;
      float v1 = acc1[r]; v1 = (v1 > 0.f) ? v1 : 0.01f * v1;
      hid[(size_t)(lh * 8 + r) * H_ + ncol] = f2bf(v0);
      hid[(size_t)(16 + lh * 8 + r) * H_ + ncol] = f2bf(v1);
    }
  }
  __syncthreads();

  // GEMM2: [32 x 1024] x [1024 x 512] -> exp_out (f32)
  for (int nt = wv * 4; nt < wv * 4 + 4; ++nt) {
    v8f acc0 = {}, acc1 = {};
    const int ncol = nt * 16 + l15;
    const unsigned short* brow = B2 + (size_t)ncol * H_;
#pragma unroll 4
    for (int kc = 0; kc < H_; kc += 32) {
      v16bf bf = ld16_contig(brow + kc + lh * 16);
      acc0 = wmma_bf16(ld16_afrag(&hid[(size_t)(l15) * H_ + kc], lh), bf, acc0);
      acc1 = wmma_bf16(ld16_afrag(&hid[(size_t)(16 + l15) * H_ + kc], lh), bf, acc1);
    }
    const int rowg = e * (B_ * CAP_) + rb * 32;
#pragma unroll
    for (int r = 0; r < 8; ++r) {
      exp_out[(size_t)(rowg + lh * 8 + r) * D_ + ncol] = acc0[r];
      exp_out[(size_t)(rowg + 16 + lh * 8 + r) * D_ + ncol] = acc1[r];
    }
  }
}

// ---------------- combine + head: out = ((g1*row1 + g2*row2)@W3 + b3)@W4 + b4 --
__global__ __launch_bounds__(128) void combine_kernel(
    const int* __restrict__ r1, const int* __restrict__ r2,
    const float* __restrict__ g1, const float* __restrict__ g2,
    const float* __restrict__ exp_out,
    const float* __restrict__ W3, const float* __restrict__ b3,
    const float* __restrict__ W4, const float* __restrict__ b4,
    float* __restrict__ out) {
  __shared__ float4 red[128];
  const int bt = blockIdx.x;
  const int tid = threadIdx.x;
  const int d0 = tid * 4;
  float4 moe = {0.f, 0.f, 0.f, 0.f};
  int a1 = r1[bt], a2 = r2[bt];
  if (a1 >= 0) {
    float g = g1[bt];
    float4 v = *reinterpret_cast<const float4*>(exp_out + (size_t)a1 * D_ + d0);
    moe.x += g * v.x; moe.y += g * v.y; moe.z += g * v.z; moe.w += g * v.w;
  }
  if (a2 >= 0) {
    float g = g2[bt];
    float4 v = *reinterpret_cast<const float4*>(exp_out + (size_t)a2 * D_ + d0);
    moe.x += g * v.x; moe.y += g * v.y; moe.z += g * v.z; moe.w += g * v.w;
  }
  float4 p = {0.f, 0.f, 0.f, 0.f};
  const float* mp = &moe.x;
#pragma unroll
  for (int j = 0; j < 4; ++j) {
    float md = mp[j];
    float4 w = *reinterpret_cast<const float4*>(W3 + (size_t)(d0 + j) * 4);
    p.x += md * w.x; p.y += md * w.y; p.z += md * w.z; p.w += md * w.w;
  }
  red[tid] = p;
  __syncthreads();
  for (int s = 64; s > 0; s >>= 1) {
    if (tid < s) {
      float4 o = red[tid + s];
      red[tid].x += o.x; red[tid].y += o.y; red[tid].z += o.z; red[tid].w += o.w;
    }
    __syncthreads();
  }
  if (tid == 0) {
    float4 a = red[0];
    out[bt] = (a.x + b3[0]) * W4[0] + (a.y + b3[1]) * W4[1] +
              (a.z + b3[2]) * W4[2] + (a.w + b3[3]) * W4[3] + b4[0];
  }
}

// ---------------- host launcher ----------------
extern "C" void kernel_launch(void* const* d_in, const int* in_sizes, int n_in,
                              void* d_out, int out_size, void* d_ws, size_t ws_size,
                              hipStream_t stream) {
  const float* X     = (const float*)d_in[0];
  const float* probs = (const float*)d_in[1];
  const float* W1    = (const float*)d_in[2];
  const float* b1    = (const float*)d_in[3];
  const float* Wg    = (const float*)d_in[4];
  const float* We1   = (const float*)d_in[5];
  const float* We2   = (const float*)d_in[6];
  const float* W3    = (const float*)d_in[7];
  const float* b3    = (const float*)d_in[8];
  const float* W4    = (const float*)d_in[9];
  const float* b4    = (const float*)d_in[10];
  float* out = (float*)d_out;

  char* w = (char*)d_ws;
  auto alloc = [&](size_t bytes) -> void* {
    void* p = (void*)w;
    w += (bytes + 255) & ~(size_t)255;
    return p;
  };
  unsigned short* Xbf    = (unsigned short*)alloc((size_t)BN_ * D_ * 2);
  unsigned short* W1t    = (unsigned short*)alloc((size_t)D_ * D_ * 2);
  unsigned short* We1t   = (unsigned short*)alloc((size_t)E_ * H_ * D_ * 2);
  unsigned short* We2t   = (unsigned short*)alloc((size_t)E_ * D_ * H_ * 2);
  float*          hF     = (float*)alloc((size_t)BN_ * D_ * 4);
  unsigned short* hBf    = (unsigned short*)alloc((size_t)BN_ * D_ * 2);
  int*   idx1   = (int*)alloc((size_t)BN_ * 4);
  int*   idx2   = (int*)alloc((size_t)BN_ * 4);
  float* g1v    = (float*)alloc((size_t)BN_ * 4);
  float* g2v    = (float*)alloc((size_t)BN_ * 4);
  int*   keep2  = (int*)alloc((size_t)BN_ * 4);
  int*   slots  = (int*)alloc((size_t)SLOTS_ * 4);
  int*   r1     = (int*)alloc((size_t)BN_ * 4);
  int*   r2     = (int*)alloc((size_t)BN_ * 4);
  unsigned short* expIn  = (unsigned short*)alloc((size_t)SLOTS_ * D_ * 2);
  float*          expOut = (float*)alloc((size_t)SLOTS_ * D_ * 4);

  // 0) precision/layout prep
  cvt_f32_bf16_kernel<<<(BN_ * D_) / 1024, 256, 0, stream>>>(X, Xbf, BN_ * D_);
  transpose_cvt_kernel<<<dim3(D_ / 32, D_ / 32, 1), dim3(32, 8), 0, stream>>>(
      W1, W1t, D_, D_, 0, 0);
  transpose_cvt_kernel<<<dim3(H_ / 32, D_ / 32, E_), dim3(32, 8), 0, stream>>>(
      We1, We1t, D_, H_, (size_t)D_ * H_, (size_t)D_ * H_);
  transpose_cvt_kernel<<<dim3(D_ / 32, H_ / 32, E_), dim3(32, 8), 0, stream>>>(
      We2, We2t, H_, D_, (size_t)D_ * H_, (size_t)D_ * H_);
  // 1) linear1
  linear1_kernel<<<dim3(BN_ / 64, D_ / 128), 256, 0, stream>>>(Xbf, W1t, b1, hF, hBf);
  // 2) gating + top2
  gating_kernel<<<BN_ / 256, 256, 0, stream>>>(hF, Wg, probs, idx1, idx2, g1v, g2v, keep2);
  // 3) ordered capacity routing
  routing_kernel<<<B_ * E_, 256, 0, stream>>>(idx1, idx2, keep2, slots, r1, r2);
  // 4) dispatch gather through the async-LDS DMA path
  dispatch_kernel<<<SLOTS_ / 4, 256, 0, stream>>>(slots, hBf, expIn);
  // 5) fused expert FFN (two WMMA GEMMs, hid in LDS)
  expert_ffn_kernel<<<E_ * 32, 256, 0, stream>>>(expIn, We1t, We2t, expOut);
  // 6) combine + head
  combine_kernel<<<BN_, 128, 0, stream>>>(r1, r2, g1v, g2v, expOut, W3, b3, W4, b4, out);
  (void)in_sizes; (void)n_in; (void)out_size; (void)ws_size;
}
